// DecoderRNN_10711648436674
// MI455X (gfx1250) — compile-verified, matrix operand order checked by reading, not run on hardware
//
#include <hip/hip_runtime.h>
#include <math.h>

// ---------------------------------------------------------------------------
// DecoderRNN for MI455X (gfx1250): bf16 WMMA GEMMs, fp32 accumulate.
//   - enc @ W1a^T hoisted out of the time loop (step-invariant)
//   - all weights converted once to bf16 (fit in 192MB L2)
//   - gemm_bf16: 128x128 block tile, 8 wave32 (2x4), 64x32 per wave,
//     v_wmma_f32_16x16x32_bf16, double-buffered LDS (80B pitch, no bank conf)
//   - tile staging via GLOBAL_LOAD_ASYNC_TO_LDS_B128 + s_wait_asynccnt
// ---------------------------------------------------------------------------

typedef __attribute__((ext_vector_type(16))) __bf16        v16bf;
typedef __attribute__((ext_vector_type(8)))  float         v8f;
typedef __attribute__((ext_vector_type(4)))  unsigned int  u32x4;
typedef __attribute__((ext_vector_type(4)))  int           i32x4;

#define B_   128
#define L_   64
#define H_   512
#define V_   32000
#define TS_  31          // decode steps (T-1)

#if defined(__AMDGCN__) && __has_builtin(__builtin_amdgcn_global_load_async_to_lds_b128)
#define HAVE_ASYNC_LDS 1
#else
#define HAVE_ASYNC_LDS 0
#endif

union FragU { v16bf v; u32x4 q2[2]; };

__device__ __forceinline__ float sigmoidf_(float x) { return 1.0f / (1.0f + expf(-x)); }

// one 16B chunk global -> LDS
__device__ __forceinline__ void stage_b128(const __bf16* g, __bf16* l) {
#if HAVE_ASYNC_LDS
  void* gv = (void*)g;   // drop const, stay generic
  void* lv = (void*)l;
  auto gp = (__attribute__((address_space(1))) i32x4*)gv;
  auto lp = (__attribute__((address_space(3))) i32x4*)lv;
  __builtin_amdgcn_global_load_async_to_lds_b128(gp, lp, 0, 0);
#else
  *(u32x4*)l = *(const u32x4*)g;
#endif
}

__device__ __forceinline__ void stage_fence() {
#if HAVE_ASYNC_LDS
#if __has_builtin(__builtin_amdgcn_s_wait_asynccnt)
  __builtin_amdgcn_s_wait_asynccnt(0);
#else
  asm volatile("s_wait_asynccnt 0x0" ::: "memory");
#endif
#endif
}

// ---------------------------------------------------------------------------
// C[row,col] = act( sum_k A[row,k]*W[col,k] + bias[col] )
// A: M x K bf16 (lda), W: N x K bf16 (ldw). M,N multiples of 128; K mult of 32.
// C (f32, ldc) and/or Cb (bf16, ldcb) optional outputs. act: 0=none, 1=tanh.
// ---------------------------------------------------------------------------
__global__ __launch_bounds__(256) void gemm_bf16(
    const __bf16* __restrict__ A, int lda,
    const __bf16* __restrict__ W, int ldw,
    const float*  __restrict__ bias,
    float*        __restrict__ C,  long long ldc,
    __bf16*       __restrict__ Cb, int ldcb,
    int K, int act)
{
  __shared__ __bf16 sA[2][128 * 40];   // 40-elem (80B) pitch -> conflict-free
  __shared__ __bf16 sB[2][128 * 40];

  const int tid     = threadIdx.x;
  const int rowBase = blockIdx.y * 128;
  const int colBase = blockIdx.x * 128;
  const int w       = tid >> 5;
  const int lane    = tid & 31;
  const int waveM   = w & 1;       // 2 waves along M (64 rows each)
  const int waveN   = w >> 1;      // 4 waves along N (32 cols each)
  const int l16     = lane & 15;
  const int lh      = lane >> 4;   // lane half
  const int koff    = lh * 8;      // K chunk start per ISA 16-bit layout

  v8f acc[4][2];
  const v8f vzero = {0.f, 0.f, 0.f, 0.f, 0.f, 0.f, 0.f, 0.f};
  #pragma unroll
  for (int mi = 0; mi < 4; ++mi)
    #pragma unroll
    for (int ni = 0; ni < 2; ++ni) acc[mi][ni] = vzero;

  const int nk = K >> 5;

  // per-thread stage coordinates: two 16B chunks per tile per operand
  const int id0 = tid;                 // chunk ids 0..511 (two each)
  const int r0  = id0 >> 2,       c0 = (id0 & 3) << 3;
  const int id1 = tid + 256;
  const int r1  = id1 >> 2,       c1 = (id1 & 3) << 3;

  const __bf16* gA0 = A + (size_t)(rowBase + r0) * lda + c0;
  const __bf16* gA1 = A + (size_t)(rowBase + r1) * lda + c1;
  const __bf16* gB0 = W + (size_t)(colBase + r0) * ldw + c0;
  const __bf16* gB1 = W + (size_t)(colBase + r1) * ldw + c1;

  // prologue: tile 0
  stage_b128(gA0, &sA[0][r0 * 40 + c0]);
  stage_b128(gA1, &sA[0][r1 * 40 + c1]);
  stage_b128(gB0, &sB[0][r0 * 40 + c0]);
  stage_b128(gB1, &sB[0][r1 * 40 + c1]);
  stage_fence();
  __syncthreads();

  for (int kt = 0; kt < nk; ++kt) {
    const int p = kt & 1;
    if (kt + 1 < nk) {
      const int k0 = (kt + 1) << 5;
      const int q  = p ^ 1;
      stage_b128(gA0 + k0, &sA[q][r0 * 40 + c0]);
      stage_b128(gA1 + k0, &sA[q][r1 * 40 + c1]);
      stage_b128(gB0 + k0, &sB[q][r0 * 40 + c0]);
      stage_b128(gB1 + k0, &sB[q][r1 * 40 + c1]);
      if (kt + 2 < nk) {  // L2 prefetch of next-next tile (global_prefetch_b8)
        __builtin_prefetch(gA0 + ((kt + 2) << 5), 0, 0);
        __builtin_prefetch(gB0 + ((kt + 2) << 5), 0, 0);
      }
    }

    FragU fa[4], fb[2];
    #pragma unroll
    for (int mi = 0; mi < 4; ++mi) {
      int r = waveM * 64 + mi * 16 + l16;
      fa[mi].q2[0] = *(const u32x4*)(&sA[p][r * 40 + koff]);
      fa[mi].q2[1] = *(const u32x4*)(&sA[p][r * 40 + koff + 16]);
    }
    #pragma unroll
    for (int ni = 0; ni < 2; ++ni) {
      int r = waveN * 32 + ni * 16 + l16;
      fb[ni].q2[0] = *(const u32x4*)(&sB[p][r * 40 + koff]);
      fb[ni].q2[1] = *(const u32x4*)(&sB[p][r * 40 + koff + 16]);
    }
    #pragma unroll
    for (int mi = 0; mi < 4; ++mi)
      #pragma unroll
      for (int ni = 0; ni < 2; ++ni)
        acc[mi][ni] = __builtin_amdgcn_wmma_f32_16x16x32_bf16(
            false, fa[mi].v, false, fb[ni].v, (short)0, acc[mi][ni], false, false);

    stage_fence();     // async writes to buffer q complete before barrier
    __syncthreads();
  }

  // epilogue: C/D layout -> lane l16 = col, VGPR j -> row j + lh*8
  #pragma unroll
  for (int ni = 0; ni < 2; ++ni) {
    int col   = colBase + waveN * 32 + ni * 16 + l16;
    float bi  = bias ? bias[col] : 0.f;
    #pragma unroll
    for (int mi = 0; mi < 4; ++mi) {
      v8f a = acc[mi][ni];
      #pragma unroll
      for (int j = 0; j < 8; ++j) {
        int   row = rowBase + waveM * 64 + mi * 16 + lh * 8 + j;
        float v   = a[j] + bi;
        if (act) v = tanhf(v);
        if (C)  C[(long long)row * ldc + col] = v;
        if (Cb) Cb[(size_t)row * ldcb + col] = (__bf16)v;
      }
    }
  }
}

// ------------------------- small elementwise kernels -----------------------

__global__ __launch_bounds__(256) void f32_to_bf16_k(const float* __restrict__ in,
                                                     __bf16* __restrict__ out, int n) {
  int i = blockIdx.x * 256 + threadIdx.x;
  if (i < n) out[i] = (__bf16)in[i];
}

__global__ __launch_bounds__(256) void split_w1_k(const float* __restrict__ W1,
                                                  __bf16* __restrict__ w1a,
                                                  __bf16* __restrict__ w1b) {
  int i = blockIdx.x * 256 + threadIdx.x;   // 512*512
  int n = i >> 9, k = i & 511;
  w1a[i] = (__bf16)W1[n * 1024 + k];
  w1b[i] = (__bf16)W1[n * 1024 + 512 + k];
}

__global__ __launch_bounds__(256) void gather_embed_k(const float* __restrict__ embW,
                                                      const int* __restrict__ tgt,
                                                      __bf16* __restrict__ xs) {
  size_t i  = (size_t)blockIdx.x * 256 + threadIdx.x;   // 31*128*512
  int    h  = (int)(i & 511);
  size_t tb = i >> 9;
  int    b  = (int)(tb & 127);
  int    t  = (int)(tb >> 7);
  int    id = tgt[b * 32 + t];
  xs[i] = (__bf16)embW[(size_t)id * H_ + h];
}

__global__ __launch_bounds__(256) void init_hid_k(const float* __restrict__ eh,
                                                  float* __restrict__ hid,
                                                  __bf16* __restrict__ hidb) {
  int i = blockIdx.x * 256 + threadIdx.x;  // 128*512
  float v = eh[i];
  hid[i]  = v;
  hidb[i] = (__bf16)v;
}

// a1 = bf16(tanh(encW1 + hc[b] + b1))   (m = b*64 + l)
__global__ __launch_bounds__(256) void a1_k(const float* __restrict__ encW1,
                                            const float* __restrict__ hc,
                                            const float* __restrict__ b1,
                                            __bf16* __restrict__ a1) {
  size_t i = (size_t)blockIdx.x * 256 + threadIdx.x;  // 8192*512
  int    n = (int)(i & 511);
  size_t m = i >> 9;
  int    b = (int)(m >> 6);
  a1[i] = (__bf16)tanhf(encW1[i] + hc[(size_t)b * H_ + n] + b1[n]);
}

// per-batch: e = a3 . v ; softmax over L; ctx = alpha @ enc; build xcat=[x_t,ctx]
__global__ __launch_bounds__(256) void attn_ctx_k(const __bf16* __restrict__ a3,
                                                  const float* __restrict__ enc,
                                                  const float* __restrict__ attv,
                                                  const __bf16* __restrict__ xs_t,
                                                  __bf16* __restrict__ xcat) {
  const int b = blockIdx.x, tid = threadIdx.x;
  __shared__ float sal[L_];
  if (tid < L_) {
    const __bf16* ar = a3 + ((size_t)(b * L_ + tid)) * H_;
    float e = 0.f;
    for (int h = 0; h < H_; ++h) e += (float)ar[h] * attv[h];
    sal[tid] = e;
  }
  __syncthreads();
  if (tid == 0) {
    float m = sal[0];
    for (int l = 1; l < L_; ++l) m = fmaxf(m, sal[l]);
    float s = 0.f;
    for (int l = 0; l < L_; ++l) { sal[l] = expf(sal[l] - m); s += sal[l]; }
    float inv = 1.f / s;
    for (int l = 0; l < L_; ++l) sal[l] *= inv;
  }
  __syncthreads();
  for (int h = tid; h < H_; h += 256) {
    float c = 0.f;
    for (int l = 0; l < L_; ++l) c += sal[l] * enc[((size_t)b * L_ + l) * H_ + h];
    xcat[(size_t)b * 1024 + 512 + h] = (__bf16)c;
    xcat[(size_t)b * 1024 + h]       = xs_t[(size_t)b * H_ + h];
  }
}

__global__ __launch_bounds__(256) void gru_elem_k(const float* __restrict__ gi,
                                                  const float* __restrict__ gh,
                                                  float* __restrict__ hid,
                                                  __bf16* __restrict__ hidb) {
  int i = blockIdx.x * 256 + threadIdx.x;  // 128*512
  int h = i & 511, b = i >> 9;
  const float* gib = gi + (size_t)b * 1536;
  const float* ghb = gh + (size_t)b * 1536;
  float r  = sigmoidf_(gib[h] + ghb[h]);
  float z  = sigmoidf_(gib[512 + h] + ghb[512 + h]);
  float n  = tanhf(gib[1024 + h] + r * ghb[1024 + h]);
  float hn = (1.f - z) * n + z * hid[i];
  hid[i]  = hn;
  hidb[i] = (__bf16)hn;
}

__global__ __launch_bounds__(256) void logsoftmax_k(float* __restrict__ out,
                                                    long long ldrow, int ncols) {
  __shared__ float red[256];
  const int tid = threadIdx.x;
  float* row = out + (long long)blockIdx.x * ldrow;
  float m = -INFINITY;
  for (int i = tid; i < ncols; i += 256) m = fmaxf(m, row[i]);
  red[tid] = m; __syncthreads();
  for (int s = 128; s > 0; s >>= 1) { if (tid < s) red[tid] = fmaxf(red[tid], red[tid + s]); __syncthreads(); }
  m = red[0]; __syncthreads();
  float sum = 0.f;
  for (int i = tid; i < ncols; i += 256) sum += expf(row[i] - m);
  red[tid] = sum; __syncthreads();
  for (int s = 128; s > 0; s >>= 1) { if (tid < s) red[tid] += red[tid + s]; __syncthreads(); }
  float ls = m + logf(red[0]);
  __syncthreads();
  for (int i = tid; i < ncols; i += 256) row[i] -= ls;
}

// ---------------------------------------------------------------------------

extern "C" void kernel_launch(void* const* d_in, const int* in_sizes, int n_in,
                              void* d_out, int out_size, void* d_ws, size_t ws_size,
                              hipStream_t stream) {
  (void)in_sizes; (void)n_in; (void)out_size; (void)ws_size;
  const float* enc  = (const float*)d_in[0];
  const float* ehid = (const float*)d_in[1];
  const int*   tgt  = (const int*)  d_in[2];
  const float* embW = (const float*)d_in[3];
  const float* aW1  = (const float*)d_in[4];
  const float* ab1  = (const float*)d_in[5];
  const float* aW2  = (const float*)d_in[6];
  const float* ab2  = (const float*)d_in[7];
  const float* aW3  = (const float*)d_in[8];
  const float* ab3  = (const float*)d_in[9];
  const float* av   = (const float*)d_in[10];
  const float* gWih = (const float*)d_in[11];
  const float* gWhh = (const float*)d_in[12];
  const float* gbih = (const float*)d_in[13];
  const float* gbhh = (const float*)d_in[14];
  const float* oW   = (const float*)d_in[15];
  const float* ob   = (const float*)d_in[16];
  float* out = (float*)d_out;

  char* ws = (char*)d_ws;
  size_t off = 0;
  auto alloc = [&](size_t bytes) -> char* {
    char* p = ws + off;
    off += (bytes + 255) & ~(size_t)255;
    return p;
  };
  __bf16* w1a   = (__bf16*)alloc((size_t)H_ * H_ * 2);
  __bf16* w1b   = (__bf16*)alloc((size_t)H_ * H_ * 2);
  __bf16* w2    = (__bf16*)alloc((size_t)H_ * H_ * 2);
  __bf16* w3    = (__bf16*)alloc((size_t)H_ * H_ * 2);
  __bf16* wih   = (__bf16*)alloc((size_t)1536 * 1024 * 2);
  __bf16* whh   = (__bf16*)alloc((size_t)1536 * H_ * 2);
  __bf16* wout  = (__bf16*)alloc((size_t)V_ * H_ * 2);
  __bf16* xs    = (__bf16*)alloc((size_t)TS_ * B_ * H_ * 2);
  __bf16* encb  = (__bf16*)alloc((size_t)B_ * L_ * H_ * 2);
  float*  encW1 = (float*) alloc((size_t)B_ * L_ * H_ * 4);
  float*  hid   = (float*) alloc((size_t)B_ * H_ * 4);
  __bf16* hidb  = (__bf16*)alloc((size_t)B_ * H_ * 2);
  float*  hc    = (float*) alloc((size_t)B_ * H_ * 4);
  __bf16* a1    = (__bf16*)alloc((size_t)B_ * L_ * H_ * 2);
  __bf16* a2    = (__bf16*)alloc((size_t)B_ * L_ * H_ * 2);
  __bf16* a3    = (__bf16*)alloc((size_t)B_ * L_ * H_ * 2);
  __bf16* xcat  = (__bf16*)alloc((size_t)B_ * 1024 * 2);
  float*  gi    = (float*) alloc((size_t)B_ * 1536 * 4);
  float*  gh    = (float*) alloc((size_t)B_ * 1536 * 4);

  const int ML = B_ * L_;  // 8192

  // ---- one-time conversions / precompute ----
  split_w1_k<<<(H_ * H_) / 256, 256, 0, stream>>>(aW1, w1a, w1b);
  f32_to_bf16_k<<<(H_ * H_) / 256, 256, 0, stream>>>(aW2, w2, H_ * H_);
  f32_to_bf16_k<<<(H_ * H_) / 256, 256, 0, stream>>>(aW3, w3, H_ * H_);
  f32_to_bf16_k<<<(1536 * 1024) / 256, 256, 0, stream>>>(gWih, wih, 1536 * 1024);
  f32_to_bf16_k<<<(1536 * H_) / 256, 256, 0, stream>>>(gWhh, whh, 1536 * H_);
  f32_to_bf16_k<<<(V_ * H_) / 256, 256, 0, stream>>>(oW, wout, V_ * H_);
  f32_to_bf16_k<<<(ML * H_) / 256, 256, 0, stream>>>(enc, encb, ML * H_);
  gather_embed_k<<<(TS_ * B_ * H_) / 256, 256, 0, stream>>>(embW, tgt, xs);
  init_hid_k<<<(B_ * H_) / 256, 256, 0, stream>>>(ehid, hid, hidb);

  // encW1 = enc @ W1a^T  (step-invariant; 8192x512x512)
  gemm_bf16<<<dim3(H_ / 128, ML / 128), 256, 0, stream>>>(
      encb, H_, w1a, H_, nullptr, encW1, (long long)H_, nullptr, 0, H_, 0);

  // ---- recurrent decode ----
  for (int t = 0; t < TS_; ++t) {
    // hc = hid @ W1b^T (128x512x512)
    gemm_bf16<<<dim3(H_ / 128, 1), 256, 0, stream>>>(
        hidb, H_, w1b, H_, nullptr, hc, (long long)H_, nullptr, 0, H_, 0);
    // a1 = tanh(encW1 + hc + b1)
    a1_k<<<(ML * H_) / 256, 256, 0, stream>>>(encW1, hc, ab1, a1);
    // a2 = tanh(a1 @ W2^T + b2), a3 = tanh(a2 @ W3^T + b3)
    gemm_bf16<<<dim3(H_ / 128, ML / 128), 256, 0, stream>>>(
        a1, H_, w2, H_, ab2, nullptr, 0, a2, H_, H_, 1);
    gemm_bf16<<<dim3(H_ / 128, ML / 128), 256, 0, stream>>>(
        a2, H_, w3, H_, ab3, nullptr, 0, a3, H_, H_, 1);
    // attention weights + context + xcat = [x_t, ctx]
    attn_ctx_k<<<B_, 256, 0, stream>>>(a3, enc, av, xs + (size_t)t * B_ * H_, xcat);
    // gi = xcat @ Wih^T + bih (128x1536x1024); gh = hid @ Whh^T + bhh
    gemm_bf16<<<dim3(1536 / 128, 1), 256, 0, stream>>>(
        xcat, 1024, wih, 1024, gbih, gi, (long long)1536, nullptr, 0, 1024, 0);
    gemm_bf16<<<dim3(1536 / 128, 1), 256, 0, stream>>>(
        hidb, H_, whh, H_, gbhh, gh, (long long)1536, nullptr, 0, H_, 0);
    gru_elem_k<<<(B_ * H_) / 256, 256, 0, stream>>>(gi, gh, hid, hidb);
    // logits -> d_out[:, t, :] (ld = 31*32000), then in-place log_softmax
    gemm_bf16<<<dim3(V_ / 128, 1), 256, 0, stream>>>(
        hidb, H_, wout, H_, ob, out + (size_t)t * V_, (long long)TS_ * V_,
        nullptr, 0, H_, 0);
    logsoftmax_k<<<B_, 256, 0, stream>>>(out + (size_t)t * V_, (long long)TS_ * V_, V_);
  }
}
